// ASTDecoder_59906203844829
// MI455X (gfx1250) — matrix operand order repaired and checked by reading, not compiled, not run on hardware
//
#include <hip/hip_runtime.h>
#include <math.h>

// ---------------------------------------------------------------------------
// MI455X (gfx1250) CDNA5 implementation of the ASTDecoder forward pass.
// All matmuls use v_wmma_f32_16x16x32_bf16 (f32 accumulate).
// GEMM A-tiles are staged into LDS by the Tensor Data Mover (TENSORcnt),
// double-buffered against the WMMA compute. Attention is flash-style.
// ---------------------------------------------------------------------------

typedef __attribute__((ext_vector_type(16))) __bf16 bf16x16;
typedef __attribute__((ext_vector_type(8)))  __bf16 bf16x8;
typedef __attribute__((ext_vector_type(8)))  float  f32x8;
typedef __attribute__((ext_vector_type(4)))  unsigned int u32x4;
typedef __attribute__((ext_vector_type(8)))  int i32x8;
typedef __attribute__((ext_vector_type(4)))  int i32x4;

#define B_   4
#define S_   1024
#define M_   1024
#define D_   512
#define H_   8
#define DK_  64
#define L_   4
#define DFF_ 2048

#define GF_GELU  1
#define GF_RESID 2
#define GF_QKV   4
#define GF_OUTBF 8

#if __has_builtin(__builtin_amdgcn_tensor_load_to_lds) && \
    __has_builtin(__builtin_amdgcn_s_wait_tensorcnt)
#define HAVE_TDM 1
#else
#define HAVE_TDM 0
#endif

__device__ __forceinline__ bf16x16 cat16(bf16x8 lo, bf16x8 hi) {
  return __builtin_shufflevector(lo, hi, 0,1,2,3,4,5,6,7,8,9,10,11,12,13,14,15);
}
__device__ __forceinline__ f32x8 wmma_bf16(bf16x16 a, bf16x16 b, f32x8 c) {
  return __builtin_amdgcn_wmma_f32_16x16x32_bf16(false, a, false, b, (short)0, c,
                                                 false, false);
}

#if HAVE_TDM
// TDM: DMA a 128x32 bf16 tile (row stride K elements) from global memory into
// LDS at lds_off, inserting 4 DWORDs of padding after every 16 DWORDs (one
// 32-element row) => LDS row stride of 40 bf16, matching the fragment layout.
// Descriptor bitfields per CDNA5 ISA.
__device__ __forceinline__ void tdm_load_a_tile(const __bf16* gaddr,
                                                unsigned lds_off, int K,
                                                int Mrows) {
  unsigned long long ga = (unsigned long long)(uintptr_t)gaddr;
  u32x4 g0;
  g0[0] = 1u;                                   // count=1, user descriptor
  g0[1] = lds_off;                              // lds_addr (bytes)
  g0[2] = (unsigned)(ga & 0xFFFFFFFFu);         // global_addr[31:0]
  g0[3] = (unsigned)((ga >> 32) & 0x1FFFFFFu)   // global_addr[56:32]
          | (2u << 30);                         // type = 2 ("image")
  i32x8 g1;
  g1[0] = (1 << 16)      // data_size = 1 -> 2 bytes
        | (1 << 20)      // pad_enable
        | (3 << 22)      // pad_interval = 3 -> every 16 DWORDs (64B row)
        | (3 << 25);     // pad_amount   = 3 -> 4 DWORDs (8 bf16)
  unsigned K_u = (unsigned)K, M_u = (unsigned)Mrows;
  g1[1] = (int)((K_u & 0xFFFFu) << 16);                       // tensor_dim0 lo
  g1[2] = (int)((K_u >> 16) | ((M_u & 0xFFFFu) << 16));       // dim0 hi|dim1 lo
  g1[3] = (int)((M_u >> 16) | (32u << 16));                   // dim1 hi|tile_dim0
  g1[4] = 128;                                                // tile_dim1
  g1[5] = (int)K_u;                                           // dim0_stride lo
  g1[6] = 0;
  g1[7] = 0;
  i32x4 z4 = {0, 0, 0, 0};
#if defined(__clang_major__) && (__clang_major__ >= 23)
  i32x8 z8 = {0, 0, 0, 0, 0, 0, 0, 0};
  __builtin_amdgcn_tensor_load_to_lds(g0, g1, z4, z4, z8, 0);
#else
  __builtin_amdgcn_tensor_load_to_lds(g0, g1, z4, z4, 0);
#endif
}
#endif

// ------------------------------ elementwise --------------------------------

__global__ void cast_f32_bf16(const float* __restrict__ src,
                              __bf16* __restrict__ dst, int n) {
  int i = (blockIdx.x * 256 + threadIdx.x) * 4;
  if (i + 3 < n) {
    float4 v = *(const float4*)(src + i);
    dst[i + 0] = (__bf16)v.x;
    dst[i + 1] = (__bf16)v.y;
    dst[i + 2] = (__bf16)v.z;
    dst[i + 3] = (__bf16)v.w;
  } else {
    for (int k = i; k < n; k++) dst[k] = (__bf16)src[k];
  }
}

__global__ void copy_f32(const float* __restrict__ src,
                         float* __restrict__ dst, int n) {
  int i = (blockIdx.x * 256 + threadIdx.x) * 4;
  if (i + 3 < n) {
    *(float4*)(dst + i) = *(const float4*)(src + i);
  } else {
    for (int k = i; k < n; k++) dst[k] = src[k];
  }
}

// ------------------------------- layernorm ---------------------------------

__global__ __launch_bounds__(256) void ln_kernel(const float* __restrict__ x,
                                                 const float* __restrict__ sc,
                                                 const float* __restrict__ bi,
                                                 __bf16* __restrict__ obf,
                                                 float* __restrict__ of) {
  __shared__ float red[16];
  const int row = blockIdx.x;
  const int t = threadIdx.x;
  const float* xr = x + (size_t)row * D_;
  float v0 = xr[t], v1 = xr[t + 256];
  float s = v0 + v1;
  float q = v0 * v0 + v1 * v1;
  for (int o = 16; o >= 1; o >>= 1) {
    s += __shfl_xor(s, o, 32);
    q += __shfl_xor(q, o, 32);
  }
  int wid = t >> 5, lane = t & 31;
  if (lane == 0) { red[wid] = s; red[8 + wid] = q; }
  __syncthreads();
  s = 0.f; q = 0.f;
  #pragma unroll
  for (int i = 0; i < 8; i++) { s += red[i]; q += red[8 + i]; }
  float mu = s * (1.f / D_);
  float var = q * (1.f / D_) - mu * mu;
  float rs = rsqrtf(var + 1e-5f);
  float o0 = (v0 - mu) * rs * sc[t] + bi[t];
  float o1 = (v1 - mu) * rs * sc[t + 256] + bi[t + 256];
  if (obf) {
    obf[(size_t)row * D_ + t] = (__bf16)o0;
    obf[(size_t)row * D_ + t + 256] = (__bf16)o1;
  } else {
    of[(size_t)row * D_ + t] = o0;
    of[(size_t)row * D_ + t + 256] = o1;
  }
}

// --------------------------------- GEMM ------------------------------------
// C[M x N] = A[M x K](bf16) * W[K x N](bf16) + bias, fused epilogues.
// Block tile 128x128, 8 waves of 64x32, K-step 32 (one WMMA K per step).
// Double-buffered LDS; A tiles DMA'd by the Tensor Data Mover (wave 0),
// B tiles register-prefetched one step ahead and transposed to n-major.

__global__ __launch_bounds__(256) void gemm_kernel(
    const __bf16* __restrict__ A0, const __bf16* __restrict__ A1,
    const __bf16* __restrict__ W, const float* __restrict__ bias,
    float* __restrict__ resid, __bf16* __restrict__ outN,
    __bf16* __restrict__ outT, int Mrows, int Ntot, int K, int Srows,
    int flags) {
  __shared__ __align__(16) __bf16 sA[2][128 * 40];
  __shared__ __align__(16) __bf16 sB[2][128 * 40];
  const int tid = threadIdx.x;
  const int lane = tid & 31, wid = tid >> 5;
  const int ln = lane & 15, hs = lane >> 4;
  const int wm = wid & 1, wn = wid >> 1;
  const int z = blockIdx.z;
  const __bf16* A = (flags & GF_QKV) ? (z == 0 ? A0 : A1) : A0;
  const __bf16* Wp = W + (size_t)z * K * Ntot;
  const float* bp = bias ? (bias + (size_t)z * Ntot) : nullptr;
  const int row0 = blockIdx.x * 128;
  const int n0 = blockIdx.y * 128;

  f32x8 acc[4][2];
  #pragma unroll
  for (int f = 0; f < 4; f++)
    #pragma unroll
    for (int g = 0; g < 2; g++) {
      float bv = bp ? bp[n0 + wn * 32 + g * 16 + ln] : 0.f;
      #pragma unroll
      for (int j = 0; j < 8; j++) acc[f][g][j] = bv;
    }

  // B staging geometry: thread covers rows (kr, kr+16) of the 32x128 tile,
  // 8 consecutive n each; transposed into sB[n][k] (stride 40).
  const int bkr = tid >> 4;          // 0..15
  const int bnn = (tid & 15) * 8;    // 0..120
  const __bf16* bptr = Wp + (size_t)bkr * Ntot + n0 + bnn;
  const size_t bstep = (size_t)32 * Ntot;

  const int nk = K >> 5;

#if !HAVE_TDM
  const __bf16* aptr = A + (size_t)(row0 + (tid >> 2)) * K + (tid & 3) * 8;
  const __bf16* aptr2 = aptr + (size_t)64 * K;  // rows 64..127
#endif

  // ---- stage tile 0 ----
#if HAVE_TDM
  if (wid == 0)
    tdm_load_a_tile(A + (size_t)row0 * K, (unsigned)(uintptr_t)&sA[0][0], K,
                    Mrows);
#else
  {
    bf16x8 a0 = *(const bf16x8*)aptr;
    bf16x8 a1 = *(const bf16x8*)aptr2;
    *(bf16x8*)&sA[0][(tid >> 2) * 40 + (tid & 3) * 8] = a0;
    *(bf16x8*)&sA[0][((tid >> 2) + 64) * 40 + (tid & 3) * 8] = a1;
  }
#endif
  {
    bf16x8 b0 = *(const bf16x8*)bptr;
    bf16x8 b1 = *(const bf16x8*)(bptr + (size_t)16 * Ntot);
    #pragma unroll
    for (int e = 0; e < 8; e++) {
      sB[0][(bnn + e) * 40 + bkr] = b0[e];
      sB[0][(bnn + e) * 40 + bkr + 16] = b1[e];
    }
  }

  for (int kt = 0; kt < nk; kt++) {
    const int buf = kt & 1;
#if HAVE_TDM
    if (wid == 0) __builtin_amdgcn_s_wait_tensorcnt(0);
#endif
    __syncthreads();  // tile kt (A via TDM, B via ds stores) now visible

    // ---- prefetch tile kt+1 while computing kt ----
    bf16x8 nb0, nb1;
    const bool has_next = (kt + 1 < nk);
    if (has_next) {
#if HAVE_TDM
      if (wid == 0)
        tdm_load_a_tile(A + (size_t)row0 * K + (kt + 1) * 32,
                        (unsigned)(uintptr_t)&sA[buf ^ 1][0], K, Mrows);
#endif
      const __bf16* bp2 = bptr + (size_t)(kt + 1) * bstep;
      nb0 = *(const bf16x8*)bp2;
      nb1 = *(const bf16x8*)(bp2 + (size_t)16 * Ntot);
    }

    // ---- compute tile kt: 12 ds_load_b128 + 8 WMMA per wave ----
    bf16x16 af[4], bfr[2];
    #pragma unroll
    for (int g = 0; g < 2; g++) {
      int n = wn * 32 + g * 16 + ln;
      bfr[g] = cat16(*(const bf16x8*)&sB[buf][n * 40 + 8 * hs],
                     *(const bf16x8*)&sB[buf][n * 40 + 16 + 8 * hs]);
    }
    #pragma unroll
    for (int f = 0; f < 4; f++) {
      int r = wm * 64 + f * 16 + ln;
      af[f] = cat16(*(const bf16x8*)&sA[buf][r * 40 + 8 * hs],
                    *(const bf16x8*)&sA[buf][r * 40 + 16 + 8 * hs]);
    }
    #pragma unroll
    for (int f = 0; f < 4; f++)
      #pragma unroll
      for (int g = 0; g < 2; g++)
        acc[f][g] = wmma_bf16(af[f], bfr[g], acc[f][g]);

    // ---- drain prefetched B into the other buffer ----
    if (has_next) {
#if !HAVE_TDM
      const __bf16* ap2 = aptr + (size_t)(kt + 1) * 32;
      const __bf16* ap3 = aptr2 + (size_t)(kt + 1) * 32;
      bf16x8 a0 = *(const bf16x8*)ap2;
      bf16x8 a1 = *(const bf16x8*)ap3;
      *(bf16x8*)&sA[buf ^ 1][(tid >> 2) * 40 + (tid & 3) * 8] = a0;
      *(bf16x8*)&sA[buf ^ 1][((tid >> 2) + 64) * 40 + (tid & 3) * 8] = a1;
#endif
      #pragma unroll
      for (int e = 0; e < 8; e++) {
        sB[buf ^ 1][(bnn + e) * 40 + bkr] = nb0[e];
        sB[buf ^ 1][(bnn + e) * 40 + bkr + 16] = nb1[e];
      }
    }
  }

  // epilogue: C layout => lane holds column n, rows j + 8*hs
  #pragma unroll
  for (int f = 0; f < 4; f++) {
    #pragma unroll
    for (int g = 0; g < 2; g++) {
      int n = n0 + wn * 32 + g * 16 + ln;
      #pragma unroll
      for (int j = 0; j < 8; j++) {
        int r = row0 + wm * 64 + f * 16 + j + 8 * hs;
        float v = acc[f][g][j];
        if (flags & GF_GELU) {
          float u = v;
          v = 0.5f * u *
              (1.f + tanhf(0.7978845608028654f * (u + 0.044715f * u * u * u)));
        }
        if (flags & GF_QKV) {
          if (z < 2) {  // Q (z=0) and K (z=1): natural [s][h][dk] bf16
            outN[(size_t)z * Mrows * Ntot + (size_t)r * Ntot + n] = (__bf16)v;
          } else {      // V: transposed [b][h][dk][s] for contiguous AV B-frags
            int b = r / Srows, sI = r - b * Srows;
            int h = n >> 6, dk = n & 63;
            outT[((size_t)(b * H_ + h) * DK_ + dk) * Srows + sI] = (__bf16)v;
          }
        } else {
          if (flags & GF_RESID) resid[(size_t)r * Ntot + n] += v;
          if (flags & GF_OUTBF) outN[(size_t)r * Ntot + n] = (__bf16)v;
        }
      }
    }
  }
}

// ----------------------------- flash attention -----------------------------
// Each wave owns a 16-row query tile; 8 waves/block cover 128 query rows.
// Per 32-key block: QK^T (4 WMMA), online softmax (shfl_xor over 16-lane
// halves, matching the C-layout row split), P transposed through per-wave LDS
// (same-wave ordering via s_wait_dscnt), then P*V (4 WMMA).

__global__ __launch_bounds__(256) void flash_kernel(
    const __bf16* __restrict__ Q, const __bf16* __restrict__ Kb,
    const __bf16* __restrict__ VT, __bf16* __restrict__ O, int Sq, int Sk,
    int causal) {
  __shared__ __align__(16) __bf16 sP[8 * 16 * 40];
  const int tid = threadIdx.x;
  const int lane = tid & 31, wid = tid >> 5;
  const int ln = lane & 15, hs = lane >> 4;
  const int b = blockIdx.y >> 3, h = blockIdx.y & 7;
  const int q0 = blockIdx.x * 128 + wid * 16;
  __bf16* wP = &sP[wid * 16 * 40];

  bf16x16 qa[2];
  const __bf16* qrow = Q + ((size_t)(b * Sq + q0 + ln)) * D_ + h * DK_;
  #pragma unroll
  for (int c = 0; c < 2; c++)
    qa[c] = cat16(*(const bf16x8*)(qrow + 32 * c + 8 * hs),
                  *(const bf16x8*)(qrow + 32 * c + 16 + 8 * hs));

  float m_i[8], l_i[8];
  f32x8 oacc[4];
  #pragma unroll
  for (int j = 0; j < 8; j++) { m_i[j] = -3.0e38f; l_i[j] = 0.f; }
  #pragma unroll
  for (int c = 0; c < 4; c++)
    #pragma unroll
    for (int j = 0; j < 8; j++) oacc[c][j] = 0.f;

  // running pointers: K rows for subtile t, V^T rows per dk chunk
  const __bf16* kp[2];
  kp[0] = Kb + ((size_t)(b * Sk + ln)) * D_ + h * DK_;
  kp[1] = kp[0] + (size_t)16 * D_;
  const __bf16* vp = VT + ((size_t)((b * H_ + h) * DK_ + ln)) * Sk + 16 * hs;

  const int nkb = causal ? ((q0 + 15) / 32 + 1) : (Sk >> 5);
  for (int kb = 0; kb < nkb; kb++) {
    const int key0 = kb * 32;
    f32x8 sc[2];
    #pragma unroll
    for (int t = 0; t < 2; t++) {
      const __bf16* krow = kp[t] + (size_t)key0 * D_;
      f32x8 a;
      #pragma unroll
      for (int j = 0; j < 8; j++) a[j] = 0.f;
      #pragma unroll
      for (int c = 0; c < 2; c++) {
        bf16x16 kf = cat16(*(const bf16x8*)(krow + 32 * c + 8 * hs),
                           *(const bf16x8*)(krow + 32 * c + 16 + 8 * hs));
        a = wmma_bf16(qa[c], kf, a);
      }
      #pragma unroll
      for (int j = 0; j < 8; j++) a[j] *= 0.125f;  // 1/sqrt(DK)
      if (causal && (key0 + 16 * t + 15 > q0)) {
        const int key = key0 + 16 * t + ln;
        #pragma unroll
        for (int j = 0; j < 8; j++) {
          int qg = q0 + j + 8 * hs;
          if (key > qg) a[j] = -1.0e30f;
        }
      }
      sc[t] = a;
    }
    // online softmax, rows live in 16-lane halves
    float corr[8];
    #pragma unroll
    for (int j = 0; j < 8; j++) {
      float rm = fmaxf(sc[0][j], sc[1][j]);
      rm = fmaxf(rm, __shfl_xor(rm, 1, 32));
      rm = fmaxf(rm, __shfl_xor(rm, 2, 32));
      rm = fmaxf(rm, __shfl_xor(rm, 4, 32));
      rm = fmaxf(rm, __shfl_xor(rm, 8, 32));
      float mn = fmaxf(m_i[j], rm);
      float cr = __expf(m_i[j] - mn);
      float p0 = __expf(sc[0][j] - mn);
      float p1 = __expf(sc[1][j] - mn);
      float rs = p0 + p1;
      rs += __shfl_xor(rs, 1, 32);
      rs += __shfl_xor(rs, 2, 32);
      rs += __shfl_xor(rs, 4, 32);
      rs += __shfl_xor(rs, 8, 32);
      l_i[j] = l_i[j] * cr + rs;
      m_i[j] = mn;
      corr[j] = cr;
      sc[0][j] = p0;
      sc[1][j] = p1;
    }
    #pragma unroll
    for (int c = 0; c < 4; c++)
      #pragma unroll
      for (int j = 0; j < 8; j++) oacc[c][j] *= corr[j];

    // transpose P (C layout -> A layout) through per-wave LDS
    #pragma unroll
    for (int t = 0; t < 2; t++)
      #pragma unroll
      for (int j = 0; j < 8; j++)
        wP[(j + 8 * hs) * 40 + 16 * t + ln] = (__bf16)sc[t][j];
    asm volatile("s_wait_dscnt 0" ::: "memory");  // same-wave DS ordering
    bf16x16 pa = cat16(*(const bf16x8*)&wP[ln * 40 + 16 * hs],
                       *(const bf16x8*)&wP[ln * 40 + 16 * hs + 8]);

    // P * V : V pre-transposed [h][dk][s] -> contiguous B fragments
    #pragma unroll
    for (int c = 0; c < 4; c++) {
      const __bf16* vrow = vp + (size_t)(16 * c) * Sk + key0;
      bf16x16 vf = cat16(*(const bf16x8*)vrow, *(const bf16x8*)(vrow + 8));
      oacc[c] = wmma_bf16(pa, vf, oacc[c]);
    }
  }

  float inv[8];
  #pragma unroll
  for (int j = 0; j < 8; j++) inv[j] = 1.f / l_i[j];
  #pragma unroll
  for (int c = 0; c < 4; c++)
    #pragma unroll
    for (int j = 0; j < 8; j++) {
      int sI = q0 + j + 8 * hs;
      O[((size_t)(b * Sq + sI)) * D_ + h * DK_ + 16 * c + ln] =
          (__bf16)(oacc[c][j] * inv[j]);
    }
}

// ------------------------------ orchestration ------------------------------

extern "C" void kernel_launch(void* const* d_in, const int* in_sizes, int n_in,
                              void* d_out, int out_size, void* d_ws,
                              size_t ws_size, hipStream_t stream) {
  (void)in_sizes; (void)n_in; (void)out_size; (void)ws_size;
  const float* tgt    = (const float*)d_in[0];
  const float* lmemf  = (const float*)d_in[1];
  const float* rmemf  = (const float*)d_in[2];
  const float* Wqkv_s = (const float*)d_in[3];
  const float* bqkv_s = (const float*)d_in[4];
  const float* Wo_s   = (const float*)d_in[5];
  const float* bo_s   = (const float*)d_in[6];
  const float* Wqkv_l = (const float*)d_in[7];
  const float* bqkv_l = (const float*)d_in[8];
  const float* Wo_l   = (const float*)d_in[9];
  const float* bo_l   = (const float*)d_in[10];
  const float* Wqkv_r = (const float*)d_in[11];
  const float* bqkv_r = (const float*)d_in[12];
  const float* Wo_r   = (const float*)d_in[13];
  const float* bo_r   = (const float*)d_in[14];
  const float* W1     = (const float*)d_in[15];
  const float* b1     = (const float*)d_in[16];
  const float* W2     = (const float*)d_in[17];
  const float* b2     = (const float*)d_in[18];
  const float* lns    = (const float*)d_in[19];
  const float* lnb    = (const float*)d_in[20];
  const float* fns    = (const float*)d_in[21];
  const float* fnb    = (const float*)d_in[22];
  // d_in[23] tgt_mask: causal mask recomputed in-kernel; [24],[25] unused.

  char* ws = (char*)d_ws;
  size_t off = 0;
  auto alloc = [&](size_t bytes) -> char* {
    char* p = ws + off;
    off += (bytes + 255) & ~(size_t)255;
    return p;
  };
  const size_t R = (size_t)B_ * S_;  // 4096 rows (B*S == B*M)
  float*  xbuf  = (float*)alloc(R * D_ * 4);
  __bf16* hbuf  = (__bf16*)alloc(R * D_ * 2);
  __bf16* qkbuf = (__bf16*)alloc(2 * R * D_ * 2);  // Q then K, contiguous
  __bf16* vtbuf = (__bf16*)alloc(R * D_ * 2);
  __bf16* aobuf = (__bf16*)alloc(R * D_ * 2);
  __bf16* midbuf= (__bf16*)alloc(R * DFF_ * 2);
  __bf16* lmem  = (__bf16*)alloc(R * D_ * 2);
  __bf16* rmem  = (__bf16*)alloc(R * D_ * 2);
  __bf16* wqkvb = (__bf16*)alloc((size_t)3 * D_ * D_ * 2);
  __bf16* wob   = (__bf16*)alloc((size_t)D_ * D_ * 2);
  __bf16* w1b   = (__bf16*)alloc((size_t)D_ * DFF_ * 2);
  __bf16* w2b   = (__bf16*)alloc((size_t)DFF_ * D_ * 2);

  auto cast = [&](const float* s, __bf16* dst, size_t n) {
    cast_f32_bf16<<<dim3((unsigned)((n + 1023) / 1024)), 256, 0, stream>>>(
        s, dst, (int)n);
  };

  copy_f32<<<dim3((unsigned)(R * D_ / 1024)), 256, 0, stream>>>(
      tgt, xbuf, (int)(R * D_));
  cast(lmemf, lmem, R * D_);
  cast(rmemf, rmem, R * D_);

  for (int l = 0; l < L_; l++) {
    struct Att {
      const float *Wqkv, *bqkv, *Wo, *bo;
      const __bf16* mem;
      int causal;
    };
    Att atts[3] = {
        {Wqkv_s + (size_t)l * 3 * D_ * D_, bqkv_s + (size_t)l * 3 * D_,
         Wo_s + (size_t)l * D_ * D_, bo_s + (size_t)l * D_, hbuf, 1},
        {Wqkv_l + (size_t)l * 3 * D_ * D_, bqkv_l + (size_t)l * 3 * D_,
         Wo_l + (size_t)l * D_ * D_, bo_l + (size_t)l * D_, lmem, 0},
        {Wqkv_r + (size_t)l * 3 * D_ * D_, bqkv_r + (size_t)l * 3 * D_,
         Wo_r + (size_t)l * D_ * D_, bo_r + (size_t)l * D_, rmem, 0},
    };
    for (int a = 0; a < 3; a++) {
      ln_kernel<<<dim3((unsigned)R), 256, 0, stream>>>(
          xbuf, lns + ((size_t)l * 4 + a) * D_, lnb + ((size_t)l * 4 + a) * D_,
          hbuf, nullptr);
      cast(atts[a].Wqkv, wqkvb, (size_t)3 * D_ * D_);
      cast(atts[a].Wo, wob, (size_t)D_ * D_);
      // fused Q/K/V projection (z = 0,1,2); V written transposed
      gemm_kernel<<<dim3((unsigned)(R / 128), D_ / 128, 3), 256, 0, stream>>>(
          hbuf, atts[a].mem, wqkvb, atts[a].bqkv, nullptr, qkbuf, vtbuf,
          (int)R, D_, D_, (a == 0 ? S_ : M_), GF_QKV);
      flash_kernel<<<dim3(S_ / 128, B_ * H_), 256, 0, stream>>>(
          qkbuf, qkbuf + R * D_, vtbuf, aobuf, S_, (a == 0 ? S_ : M_),
          atts[a].causal);
      // output projection + residual into fp32 stream
      gemm_kernel<<<dim3((unsigned)(R / 128), D_ / 128, 1), 256, 0, stream>>>(
          aobuf, nullptr, wob, atts[a].bo, xbuf, nullptr, nullptr, (int)R, D_,
          D_, S_, GF_RESID);
    }
    // FFN
    ln_kernel<<<dim3((unsigned)R), 256, 0, stream>>>(
        xbuf, lns + ((size_t)l * 4 + 3) * D_, lnb + ((size_t)l * 4 + 3) * D_,
        hbuf, nullptr);
    cast(W1 + (size_t)l * D_ * DFF_, w1b, (size_t)D_ * DFF_);
    cast(W2 + (size_t)l * DFF_ * D_, w2b, (size_t)DFF_ * D_);
    gemm_kernel<<<dim3((unsigned)(R / 128), DFF_ / 128, 1), 256, 0, stream>>>(
        hbuf, nullptr, w1b, b1 + (size_t)l * DFF_, nullptr, midbuf, nullptr,
        (int)R, DFF_, D_, S_, GF_OUTBF | GF_GELU);
    gemm_kernel<<<dim3((unsigned)(R / 128), D_ / 128, 1), 256, 0, stream>>>(
        midbuf, nullptr, w2b, b2 + (size_t)l * D_, xbuf, nullptr, nullptr,
        (int)R, D_, DFF_, S_, GF_RESID);
  }
  ln_kernel<<<dim3((unsigned)R), 256, 0, stream>>>(xbuf, fns, fnb, nullptr,
                                                   (float*)d_out);
}
